// KernelTransformer_12893491823374
// MI455X (gfx1250) — matrix operand order repaired and checked by reading, not compiled, not run on hardware
//
#include <hip/hip_runtime.h>
#include <hip/hip_bf16.h>
#include <math.h>

typedef __attribute__((ext_vector_type(16))) _Float16 v16h;
typedef __attribute__((ext_vector_type(8)))  _Float16 h8;
typedef __attribute__((ext_vector_type(8)))  float    v8f;

#define C_DIM 256
#define HH    17
#define WW    16
#define NPIX  (16 * HH * WW)   // 4352 pixels (B * 17 * 16)

// ---------------------------------------------------------------------------
// WMMA helpers (gfx1250, wave32). f16 in, f32 accumulate.
// A (16x32 MxK): lane r=L&15 is row M=r; 16 contiguous K = (L>>4)*16+0..15
// B (32x16 KxN): lane holds column N=L&15; 16 contiguous K = (L>>4)*16+0..15
// ---------------------------------------------------------------------------
__device__ __forceinline__ v16h load_fragA(const _Float16* row, int kc) {
  int hi = (threadIdx.x >> 4) & 1;
  const _Float16* p = row + kc + hi * 16;
  h8 lo = *(const h8*)p;
  h8 hv = *(const h8*)(p + 8);
  v16h a;
#pragma unroll
  for (int i = 0; i < 8; ++i) { a[i] = lo[i]; a[8 + i] = hv[i]; }
  return a;
}

__device__ __forceinline__ v16h load_fragB(const _Float16* row, int kc) {
  int hi = (threadIdx.x >> 4) & 1;
  const _Float16* p = row + kc + hi * 16;
  h8 lo = *(const h8*)p;
  h8 hv = *(const h8*)(p + 8);
  v16h b;
#pragma unroll
  for (int i = 0; i < 8; ++i) { b[i] = lo[i]; b[8 + i] = hv[i]; }
  return b;
}

__device__ __forceinline__ v8f wmma32(v16h a, v16h b, v8f c) {
  return __builtin_amdgcn_wmma_f32_16x16x32_f16(false, a, false, b, (short)0, c,
                                                false, false);
}

// ---------------------------------------------------------------------------
// f32 -> f16 weight conversion (with zero padding past nsrc)
// ---------------------------------------------------------------------------
__global__ void cvt_f16_k(const float* __restrict__ s, _Float16* __restrict__ d,
                          int nsrc, int ntot) {
  int i = blockIdx.x * 256 + threadIdx.x;
  if (i < ntot) d[i] = (i < nsrc) ? (_Float16)s[i] : (_Float16)0.f;
}

// gather the 16 cls pixels (b,0,0,:) into a dense 16x256 f16 matrix
__global__ void cvt_cls_k(const float* __restrict__ hbuf, _Float16* __restrict__ d) {
  int i = blockIdx.x * 256 + threadIdx.x;           // 16*256
  int b = i >> 8, c = i & 255;
  d[i] = (_Float16)hbuf[(size_t)b * HH * WW * C_DIM + c];
}

// ---------------------------------------------------------------------------
// Patch embed (2x2 conv stride 2) + 2D pos emb + cls row prepend
// one block per output pixel, thread = channel
// ---------------------------------------------------------------------------
__global__ void patch_embed_k(const float* __restrict__ x, const float* __restrict__ pw,
                              const float* __restrict__ pb, const float* __restrict__ rowe,
                              const float* __restrict__ cole, const float* __restrict__ clst,
                              float* __restrict__ hbuf) {
  int pix = blockIdx.x;                // b*17*16 + hh*16 + ww
  int c = threadIdx.x;
  int b = pix / (HH * WW);
  int rem = pix % (HH * WW);
  int hh = rem / WW, ww = rem % WW;
  float v;
  if (hh == 0) {
    v = clst[c];
  } else {
    int gh = hh - 1, gw = ww;
    float s = pb[c];
#pragma unroll
    for (int ic = 0; ic < 3; ++ic)
#pragma unroll
      for (int kh = 0; kh < 2; ++kh)
#pragma unroll
        for (int kw = 0; kw < 2; ++kw)
          s += x[((b * 3 + ic) * 32 + (2 * gh + kh)) * 32 + (2 * gw + kw)] *
               pw[((c * 3 + ic) * 2 + kh) * 2 + kw];
    s += (c < 128) ? rowe[gh * 128 + c] : cole[gw * 128 + (c - 128)];
    v = s;
  }
  hbuf[(size_t)pix * C_DIM + c] = v;
}

// ---------------------------------------------------------------------------
// LayerNorm over C=256, output f16. One wave per pixel, 8 pixels per block.
// ---------------------------------------------------------------------------
__global__ void ln_k(const float* __restrict__ hbuf, const float* __restrict__ w,
                     const float* __restrict__ b, _Float16* __restrict__ y) {
  int pix = blockIdx.x * 8 + (threadIdx.x >> 5);
  int lane = threadIdx.x & 31;
  const float* r = hbuf + (size_t)pix * C_DIM;
  float vals[8];
  float s = 0.f, s2 = 0.f;
#pragma unroll
  for (int i = 0; i < 8; ++i) {
    float v = r[lane + i * 32];
    vals[i] = v; s += v; s2 += v * v;
  }
#pragma unroll
  for (int off = 16; off; off >>= 1) {
    s  += __shfl_xor(s,  off, 32);
    s2 += __shfl_xor(s2, off, 32);
  }
  float mu = s * (1.f / 256.f);
  float var = s2 * (1.f / 256.f) - mu * mu;
  float inv = rsqrtf(var + 1e-5f);
  _Float16* yo = y + (size_t)pix * C_DIM;
#pragma unroll
  for (int i = 0; i < 8; ++i) {
    int c = lane + i * 32;
    yo[c] = (_Float16)((vals[i] - mu) * inv * w[c] + b[c]);
  }
}

// ---------------------------------------------------------------------------
// Generic WMMA GEMM: out[m,n] = sum_k A[m,k] * W[n,k]  (+bias, +epilogue)
// One wave computes a 16 x (NT*16) tile: A fragment loaded once per K-step and
// reused across NT B fragments / WMMAs (raises WMMA density per byte; the
// workload is L2-resident and compute-bound).  grid = (M/16, N/(NT*16)).
// MODE 0: out f16 (QKV)      MODE 1: out f16 = gelu(acc+bias)   (MLP1)
// MODE 2: out f32 += acc+bias (MLP2 residual)  MODE 3: out f32 = acc+bias, n<realN
// ---------------------------------------------------------------------------
template <int MODE, int NT>
__global__ void __launch_bounds__(32)
gemm_k(const _Float16* __restrict__ A, int lda,
       const _Float16* __restrict__ W, int ldw,
       const float* __restrict__ bias,
       _Float16* __restrict__ o16, float* __restrict__ o32,
       int ldo, int K, int realN) {
  int lane = threadIdx.x & 31;
  int rlo = lane & 15, hi = lane >> 4;
  const _Float16* arow = A + (size_t)(blockIdx.x * 16 + rlo) * lda;
  const _Float16* wrow[NT];
#pragma unroll
  for (int t = 0; t < NT; ++t)
    wrow[t] = W + (size_t)(blockIdx.y * (NT * 16) + t * 16 + rlo) * ldw;

  v8f acc[NT];
#pragma unroll
  for (int t = 0; t < NT; ++t) acc[t] = (v8f){};

  for (int kc = 0; kc < K; kc += 32) {
    v16h a = load_fragA(arow, kc);
#pragma unroll
    for (int t = 0; t < NT; ++t)
      acc[t] = wmma32(a, load_fragB(wrow[t], kc), acc[t]);
  }

#pragma unroll
  for (int t = 0; t < NT; ++t) {
    int n = blockIdx.y * (NT * 16) + t * 16 + rlo;
    float bv = (MODE == 0) ? 0.f : bias[n];
#pragma unroll
    for (int r = 0; r < 8; ++r) {
      int m = blockIdx.x * 16 + hi * 8 + r;
      float v = acc[t][r] + bv;
      if (MODE == 0) {
        o16[(size_t)m * ldo + n] = (_Float16)v;
      } else if (MODE == 1) {
        o16[(size_t)m * ldo + n] =
            (_Float16)(0.5f * v * (1.f + erff(v * 0.70710678118654752f)));
      } else if (MODE == 2) {
        o32[(size_t)m * ldo + n] += v;
      } else {
        if (n < realN) o32[(size_t)m * ldo + n] = v;
      }
    }
  }
}

// ---------------------------------------------------------------------------
// Sliding-window attention, one workgroup (8 waves) per window.
//   S = scale * Q K^T  (WMMA, f32 in LDS)  -> row softmax (packed f16 in place)
//   O = P V (WMMA)     -> overlap-add fold into hbuf via f32 atomics
// qkv layout: per pixel 768 f16 = [Q 0..255 | K 256..511 | V 512..767]
// ---------------------------------------------------------------------------
__global__ void attn_k(const _Float16* __restrict__ qkv, float* __restrict__ hbuf,
                       int k, int Lh, int Lw, float scale) {
  extern __shared__ float S[];        // nt*nt f32, later nt rows of f16 P
  __shared__ int pixoff[256];         // token -> pixel index (within hbuf/qkv)

  int win = blockIdx.x;
  int b = win / (Lh * Lw);
  int rem = win % (Lh * Lw);
  int lh = rem / Lw, lw = rem % Lw;
  int nt = k * k;
  int ntile = nt >> 4;
  int ntp = (nt + 31) & ~31;          // K padded to a multiple of 32

  int lane = threadIdx.x & 31;
  int wid = threadIdx.x >> 5;
  int nw = blockDim.x >> 5;
  int rlo = lane & 15, hi = lane >> 4;

  for (int t = threadIdx.x; t < nt; t += blockDim.x)
    pixoff[t] = (b * HH + lh + t / k) * WW + lw + t % k;
  __syncthreads();

  // ---- Phase 1: S = scale * Q K^T ----
  for (int t = wid; t < ntile * ntile; t += nw) {
    int mi = t / ntile, ni = t % ntile;
    const _Float16* qrow = qkv + (size_t)pixoff[mi * 16 + rlo] * 768;        // Q
    const _Float16* krow = qkv + (size_t)pixoff[ni * 16 + rlo] * 768 + 256;  // K
    v8f acc = {};
    for (int kc = 0; kc < C_DIM; kc += 32)
      acc = wmma32(load_fragA(qrow, kc), load_fragB(krow, kc), acc);
#pragma unroll
    for (int r = 0; r < 8; ++r)
      S[(size_t)(mi * 16 + hi * 8 + r) * nt + ni * 16 + rlo] = acc[r] * scale;
  }
  __syncthreads();

  // ---- Phase 2: row softmax; pack P as f16 into the front of each f32 row ----
  for (int row = threadIdx.x; row < nt; row += blockDim.x) {
    float* sr = S + (size_t)row * nt;
    _Float16* pr = (_Float16*)sr;
    float mx = -1e30f;
    for (int j = 0; j < nt; ++j) mx = fmaxf(mx, sr[j]);
    float sum = 0.f;
    for (int j = 0; j < nt; ++j) {      // safe in-place: write byte 2j < read byte 4j
      float e = __expf(sr[j] - mx);
      sum += e;
      pr[j] = (_Float16)e;
    }
    float inv = 1.f / sum;
    for (int j = 0; j < nt; ++j) pr[j] = (_Float16)((float)pr[j] * inv);
    for (int j = nt; j < ntp; ++j) pr[j] = (_Float16)0.f;   // zero-pad K to 32
  }
  __syncthreads();

  // ---- Phase 3: O = P V ; fold (overlap-add) into hbuf ----
  for (int t = wid; t < ntile * 16; t += nw) {
    int mi = t >> 4, ni = t & 15;                      // 16 channel tiles (C=256)
    const _Float16* prow = ((const _Float16*)S) + (size_t)(mi * 16 + rlo) * 2 * nt;
    int ncol = ni * 16 + rlo;
    v8f acc = {};
    for (int kc = 0; kc < ntp; kc += 32) {
      v16h a = load_fragA(prow, kc);
      v16h bb;
      int kb = kc + hi * 16;
#pragma unroll
      for (int j = 0; j < 16; ++j) {
        int tok = kb + j;
        if (tok >= nt) tok = 0;                        // P is zero there anyway
        bb[j] = qkv[(size_t)pixoff[tok] * 768 + 512 + ncol];   // V column gather
      }
      acc = wmma32(a, bb, acc);
    }
#pragma unroll
    for (int r = 0; r < 8; ++r) {
      int m = mi * 16 + hi * 8 + r;
      float* dst = hbuf + (size_t)pixoff[m] * C_DIM + ncol;
      __hip_atomic_fetch_add(dst, acc[r], __ATOMIC_RELAXED, __HIP_MEMORY_SCOPE_AGENT);
    }
  }
}

// ---------------------------------------------------------------------------
extern "C" void kernel_launch(void* const* d_in, const int* in_sizes, int n_in,
                              void* d_out, int out_size, void* d_ws, size_t ws_size,
                              hipStream_t stream) {
  const float* x       = (const float*)d_in[0];
  const float* patch_w = (const float*)d_in[1];
  const float* patch_b = (const float*)d_in[2];
  const float* row_emb = (const float*)d_in[3];
  const float* col_emb = (const float*)d_in[4];
  const float* cls_tok = (const float*)d_in[5];
  const float* ln1_w   = (const float*)d_in[6];
  const float* ln1_b   = (const float*)d_in[7];
  const float* qkv_w   = (const float*)d_in[8];
  const float* ln2_w   = (const float*)d_in[9];
  const float* ln2_b   = (const float*)d_in[10];
  const float* mlp_w1  = (const float*)d_in[11];
  const float* mlp_b1  = (const float*)d_in[12];
  const float* mlp_w2  = (const float*)d_in[13];
  const float* mlp_b2  = (const float*)d_in[14];
  const float* cls_w   = (const float*)d_in[15];
  const float* cls_b   = (const float*)d_in[16];
  float* out = (float*)d_out;

  char* ws = (char*)d_ws;
  size_t off = 0;
  auto alloc = [&](size_t bytes) {
    char* p = ws + off;
    off = (off + bytes + 255) & ~(size_t)255;
    return p;
  };
  float*    hbuf  = (float*)   alloc((size_t)NPIX * 256 * 4);   // activations f32
  _Float16* y16   = (_Float16*)alloc((size_t)NPIX * 256 * 2);   // LN output f16
  _Float16* qkv16 = (_Float16*)alloc((size_t)NPIX * 768 * 2);   // per-pixel QKV f16
  _Float16* hid16 = (_Float16*)alloc((size_t)NPIX * 512 * 2);   // MLP hidden f16
  _Float16* wq16  = (_Float16*)alloc((size_t)7 * 768 * 256 * 2);
  _Float16* w1h   = (_Float16*)alloc((size_t)7 * 512 * 256 * 2);
  _Float16* w2h   = (_Float16*)alloc((size_t)7 * 256 * 512 * 2);
  _Float16* cwh   = (_Float16*)alloc((size_t)1008 * 256 * 2);   // cls_w padded
  _Float16* cls16 = (_Float16*)alloc((size_t)16 * 256 * 2);

  // one-shot weight conversion to f16
  {
    int n0 = 7 * 768 * 256, n1 = 7 * 512 * 256, n2 = 7 * 256 * 512;
    cvt_f16_k<<<(n0 + 255) / 256, 256, 0, stream>>>(qkv_w,  wq16, n0, n0);
    cvt_f16_k<<<(n1 + 255) / 256, 256, 0, stream>>>(mlp_w1, w1h,  n1, n1);
    cvt_f16_k<<<(n2 + 255) / 256, 256, 0, stream>>>(mlp_w2, w2h,  n2, n2);
    cvt_f16_k<<<(1008 * 256 + 255) / 256, 256, 0, stream>>>(cls_w, cwh,
                                                            1000 * 256, 1008 * 256);
  }

  patch_embed_k<<<NPIX, 256, 0, stream>>>(x, patch_w, patch_b, row_emb, col_emb,
                                          cls_tok, hbuf);

  const int   kk_[7]  = {4, 4, 8, 8, 8, 16, 16};
  const float scl_[7] = {0.125f, 0.125f, 0.17677669529663687f, 0.17677669529663687f,
                         0.17677669529663687f, 0.25f, 0.25f};

  for (int i = 0; i < 7; ++i) {
    int k = kk_[i], Lh = HH - k + 1, Lw = WW - k + 1, nt = k * k;

    // attention branch
    ln_k<<<NPIX / 8, 256, 0, stream>>>(hbuf, ln1_w + i * 256, ln1_b + i * 256, y16);
    gemm_k<0, 4><<<dim3(NPIX / 16, 768 / 64), 32, 0, stream>>>(
        y16, 256, wq16 + (size_t)i * 768 * 256, 256, nullptr,
        qkv16, nullptr, 768, 256, 768);
    attn_k<<<16 * Lh * Lw, 256, (size_t)nt * nt * 4, stream>>>(
        qkv16, hbuf, k, Lh, Lw, scl_[i]);

    // MLP branch
    ln_k<<<NPIX / 8, 256, 0, stream>>>(hbuf, ln2_w + i * 256, ln2_b + i * 256, y16);
    gemm_k<1, 4><<<dim3(NPIX / 16, 512 / 64), 32, 0, stream>>>(
        y16, 256, w1h + (size_t)i * 512 * 256, 256, mlp_b1 + i * 512,
        hid16, nullptr, 512, 256, 512);
    gemm_k<2, 4><<<dim3(NPIX / 16, 256 / 64), 32, 0, stream>>>(
        hid16, 512, w2h + (size_t)i * 256 * 512, 512, mlp_b2 + i * 256,
        nullptr, hbuf, 256, 512, 256);
  }

  // classifier head: (16 x 256) @ (1000 x 256)^T + bias   (1008 = 21 * 48)
  cvt_cls_k<<<16, 256, 0, stream>>>(hbuf, cls16);
  gemm_k<3, 3><<<dim3(1, 21), 32, 0, stream>>>(
      cls16, 256, cwh, 256, cls_b, nullptr, out, 1000, 256, 1000);
}
